// LSTMDecoder_73710228734837
// MI455X (gfx1250) — matrix operand order repaired and checked by reading, compiled, not run on hardware
//
#include <hip/hip_runtime.h>
#include <hip/hip_bf16.h>
#include <math.h>

// ---------------------------------------------------------------------------
// Types for CDNA5 WMMA bf16 fragments
// ---------------------------------------------------------------------------
typedef __attribute__((ext_vector_type(16))) __bf16 bf16x16;
typedef __attribute__((ext_vector_type(8)))  __bf16 bf16x8;
typedef __attribute__((ext_vector_type(8)))  float  floatx8;

union bfu { bf16x16 v; bf16x8 h[2]; };

// A fragment (16x32 bf16): lane L -> row m = L%16; holds K = {16*ha..16*ha+15}
// where ha = L/16.  Two 16-byte contiguous loads.
__device__ __forceinline__ bf16x16 load_a_frag(const __bf16* Arow, int kc, int ha) {
    bfu u;
    u.h[0] = *(const bf16x8*)(Arow + kc + 16 * ha);
    u.h[1] = *(const bf16x8*)(Arow + kc + 16 * ha + 8);
    return u.v;
}

// B fragment (32x16 bf16): lane L -> column n = L%16; holds K = {16*ha..16*ha+15}.
// With weights stored row-major [N, K], that's 16 contiguous bf16 from row n.
__device__ __forceinline__ bf16x16 load_b_frag(const __bf16* Wrow, int kc, int ha) {
    bfu u;
    u.h[0] = *(const bf16x8*)(Wrow + kc + 16 * ha);
    u.h[1] = *(const bf16x8*)(Wrow + kc + 16 * ha + 8);
    return u.v;
}

__device__ __forceinline__ float sigf(float x) { return 1.0f / (1.0f + __expf(-x)); }

// ---------------------------------------------------------------------------
// Small prep kernels
// ---------------------------------------------------------------------------
__global__ void cvt_bf16_kernel(const float* __restrict__ s, __bf16* __restrict__ d, int n) {
    int i = blockIdx.x * 256 + threadIdx.x;
    if (i < n) d[i] = (__bf16)s[i];
}

__global__ void bias_sum_kernel(const float* __restrict__ a, const float* __restrict__ b,
                                float* __restrict__ d, int n) {
    int i = blockIdx.x * 256 + threadIdx.x;
    if (i < n) d[i] = a[i] + b[i];
}

__global__ void zero_f32_kernel(float* p, int n) {
    int i = blockIdx.x * 256 + threadIdx.x;
    if (i < n) p[i] = 0.0f;
}

__global__ void zero_bf16_kernel(__bf16* p, int n) {
    int i = blockIdx.x * 256 + threadIdx.x;
    if (i < n) p[i] = (__bf16)0.0f;
}

// Gather embedding rows for x = emb[:, :-1, :], ordered m = t*64 + b, bf16.
__global__ void gather_embed_kernel(const int* __restrict__ captions,
                                    const float* __restrict__ embed,
                                    __bf16* __restrict__ X) {
    int m = blockIdx.x;          // 0..4031
    int t = m >> 6;              // 0..62
    int b = m & 63;
    int tok = captions[b * 64 + t];
    const float* src = embed + (size_t)tok * 512;
    __bf16* dst = X + (size_t)m * 512;
    dst[threadIdx.x]       = (__bf16)src[threadIdx.x];
    dst[threadIdx.x + 256] = (__bf16)src[threadIdx.x + 256];
}

// ---------------------------------------------------------------------------
// Software-pipelined wave-level GEMM body: 16(M) x 64(N) per wave, K = 512.
// Double-buffers A/B fragments so next-chunk loads are issued before the
// current chunk's WMMAs (breaks the load->wait0->wmma serialization).
// Wrow[j] are the 4 weight-row base pointers (already lane-resolved rows).
// ---------------------------------------------------------------------------
__device__ __forceinline__ void gemm_wave_16x64_pipelined(
    floatx8 acc[4], const __bf16* __restrict__ Arow,
    const __bf16* __restrict__ Wrow0, const __bf16* __restrict__ Wrow1,
    const __bf16* __restrict__ Wrow2, const __bf16* __restrict__ Wrow3,
    int ha) {
    const __bf16* Wrow[4] = {Wrow0, Wrow1, Wrow2, Wrow3};
    bf16x16 a_cur = load_a_frag(Arow, 0, ha);
    bf16x16 b_cur[4];
#pragma unroll
    for (int j = 0; j < 4; ++j) b_cur[j] = load_b_frag(Wrow[j], 0, ha);

#pragma unroll 4
    for (int kc = 0; kc < 512; kc += 32) {
        // Prefetch next chunk (wrap to 0 on the last iteration; values unused)
        int kn = (kc + 32 < 512) ? (kc + 32) : 0;
        bf16x16 a_nxt = load_a_frag(Arow, kn, ha);
        bf16x16 b_nxt[4];
#pragma unroll
        for (int j = 0; j < 4; ++j) b_nxt[j] = load_b_frag(Wrow[j], kn, ha);

#pragma unroll
        for (int j = 0; j < 4; ++j)
            acc[j] = __builtin_amdgcn_wmma_f32_16x16x32_bf16(
                false, a_cur, false, b_cur[j], (short)0, acc[j], false, false);

        a_cur = a_nxt;
#pragma unroll
        for (int j = 0; j < 4; ++j) b_cur[j] = b_nxt[j];
    }
}

// ---------------------------------------------------------------------------
// GEMM: xg0 = X(4032x512) * W_ih_0^T(512x2048) + bias0.   out[m, n], row-major.
// Block: 256 threads = 8 waves. Block tile 64(M) x 128(N); wave tile 16 x 64.
// ---------------------------------------------------------------------------
__global__ __launch_bounds__(256) void gemm_xg0_kernel(
    const __bf16* __restrict__ A, const __bf16* __restrict__ W,
    const float* __restrict__ bias, float* __restrict__ out) {
    const int lane = threadIdx.x & 31;
    const int wave = threadIdx.x >> 5;
    const int ln = lane & 15, ha = lane >> 4;
    const int m0 = blockIdx.y * 64 + (wave & 3) * 16;
    const int n0 = blockIdx.x * 128 + (wave >> 2) * 64;
    const __bf16* Arow = A + (size_t)(m0 + ln) * 512;

    floatx8 acc[4];
#pragma unroll
    for (int j = 0; j < 4; ++j) {
        float bv = bias[n0 + j * 16 + ln];
#pragma unroll
        for (int r = 0; r < 8; ++r) acc[j][r] = bv;
    }

    gemm_wave_16x64_pipelined(acc, Arow,
                              W + (size_t)(n0 + 0  + ln) * 512,
                              W + (size_t)(n0 + 16 + ln) * 512,
                              W + (size_t)(n0 + 32 + ln) * 512,
                              W + (size_t)(n0 + 48 + ln) * 512, ha);

#pragma unroll
    for (int j = 0; j < 4; ++j) {
        int n = n0 + j * 16 + ln;
#pragma unroll
        for (int r = 0; r < 8; ++r) {
            int m = m0 + 8 * ha + r;
            out[(size_t)m * 2048 + n] = acc[j][r];
        }
    }
}

// ---------------------------------------------------------------------------
// Final FC: out[b,t,v] = h2[m=t*64+b, :] . fc_W[v, :] + fc_b[v]
// M=4032, N=50257 (bounds-guarded), K=512.  Same tiling, pipelined.
// ---------------------------------------------------------------------------
__global__ __launch_bounds__(256) void gemm_fc_kernel(
    const __bf16* __restrict__ A, const __bf16* __restrict__ W,
    const float* __restrict__ bias, float* __restrict__ out, int N) {
    const int lane = threadIdx.x & 31;
    const int wave = threadIdx.x >> 5;
    const int ln = lane & 15, ha = lane >> 4;
    const int m0 = blockIdx.y * 64 + (wave & 3) * 16;
    const int n0 = blockIdx.x * 128 + (wave >> 2) * 64;
    const __bf16* Arow = A + (size_t)(m0 + ln) * 512;

    int nc[4];
    floatx8 acc[4];
#pragma unroll
    for (int j = 0; j < 4; ++j) {
        int nj = n0 + j * 16 + ln;
        nc[j] = nj < N ? nj : (N - 1);       // clamp for safe loads
        float bv = bias[nc[j]];
#pragma unroll
        for (int r = 0; r < 8; ++r) acc[j][r] = bv;
    }

    gemm_wave_16x64_pipelined(acc, Arow,
                              W + (size_t)nc[0] * 512,
                              W + (size_t)nc[1] * 512,
                              W + (size_t)nc[2] * 512,
                              W + (size_t)nc[3] * 512, ha);

#pragma unroll
    for (int j = 0; j < 4; ++j) {
        int n = n0 + j * 16 + ln;
        if (n < N) {
#pragma unroll
            for (int r = 0; r < 8; ++r) {
                int m = m0 + 8 * ha + r;
                int t = m >> 6, b = m & 63;
                out[(size_t)(b * 63 + t) * N + n] = acc[j][r];
            }
        }
    }
}

// ---------------------------------------------------------------------------
// Fused LSTM step.  Grid: 8 blocks x 256 threads.  Each block owns 64
// h-columns across all 4 gates (offsets 0/512/1024/1536) so the pointwise
// update fuses on the accumulators.  Wave: mt = wave&3 (16 batch rows),
// hs = wave>>2 (32 h-cols -> 2 subtiles x 4 gates = 8 WMMA accumulators).
//   gate_init != null : accumulators start from precomputed xg0[t] (layer 0)
//   else              : accumulators start from bias (layer 1)
//   A1/W1 != null     : extra input-path GEMM (layer 1 uses h1_t)
//   A2/W2             : recurrent GEMM (h_prev @ W_hh^T)
// ---------------------------------------------------------------------------
__device__ __forceinline__ void gemm_acc8(
    floatx8 acc[4][2], const __bf16* __restrict__ A, const __bf16* __restrict__ W,
    int m0, int hb, int ln, int ha) {
    const __bf16* Arow = A + (size_t)(m0 + ln) * 512;
    for (int kc = 0; kc < 512; kc += 32) {
        bf16x16 a = load_a_frag(Arow, kc, ha);
#pragma unroll
        for (int g = 0; g < 4; ++g) {
#pragma unroll
            for (int j = 0; j < 2; ++j) {
                const __bf16* Wrow = W + (size_t)(g * 512 + hb + j * 16 + ln) * 512;
                bf16x16 bm = load_b_frag(Wrow, kc, ha);
                acc[g][j] = __builtin_amdgcn_wmma_f32_16x16x32_bf16(
                    false, a, false, bm, (short)0, acc[g][j], false, false);
            }
        }
    }
}

__global__ __launch_bounds__(256) void lstm_step_kernel(
    const float* __restrict__ gate_init, const float* __restrict__ bias,
    const __bf16* __restrict__ A1, const __bf16* __restrict__ W1,
    const __bf16* __restrict__ A2, const __bf16* __restrict__ W2,
    float* __restrict__ c_state, __bf16* __restrict__ h_out) {
    const int lane = threadIdx.x & 31;
    const int wave = threadIdx.x >> 5;
    const int ln = lane & 15, ha = lane >> 4;
    const int m0 = (wave & 3) * 16;                     // batch rows
    const int hb = blockIdx.x * 64 + (wave >> 2) * 32;  // h-column base

    floatx8 acc[4][2];
    if (gate_init) {
#pragma unroll
        for (int g = 0; g < 4; ++g)
#pragma unroll
            for (int j = 0; j < 2; ++j) {
                int col = g * 512 + hb + j * 16 + ln;
#pragma unroll
                for (int r = 0; r < 8; ++r)
                    acc[g][j][r] = gate_init[(size_t)(m0 + 8 * ha + r) * 2048 + col];
            }
    } else {
#pragma unroll
        for (int g = 0; g < 4; ++g)
#pragma unroll
            for (int j = 0; j < 2; ++j) {
                float bv = bias[g * 512 + hb + j * 16 + ln];
#pragma unroll
                for (int r = 0; r < 8; ++r) acc[g][j][r] = bv;
            }
    }

    if (A1) gemm_acc8(acc, A1, W1, m0, hb, ln, ha);   // layer-1 input path
    gemm_acc8(acc, A2, W2, m0, hb, ln, ha);           // recurrent path

    // Pointwise LSTM cell update (gate order i, f, g, o)
#pragma unroll
    for (int j = 0; j < 2; ++j) {
        int hcol = hb + j * 16 + ln;
#pragma unroll
        for (int r = 0; r < 8; ++r) {
            int m = m0 + 8 * ha + r;
            float iv = sigf(acc[0][j][r]);
            float fv = sigf(acc[1][j][r]);
            float gv = tanhf(acc[2][j][r]);
            float ov = sigf(acc[3][j][r]);
            size_t idx = (size_t)m * 512 + hcol;
            float cn = fv * c_state[idx] + iv * gv;
            c_state[idx] = cn;
            h_out[idx] = (__bf16)(ov * tanhf(cn));
        }
    }
}

// ---------------------------------------------------------------------------
// Host launcher
// ---------------------------------------------------------------------------
extern "C" void kernel_launch(void* const* d_in, const int* in_sizes, int n_in,
                              void* d_out, int out_size, void* d_ws, size_t ws_size,
                              hipStream_t stream) {
    (void)in_sizes; (void)n_in; (void)out_size; (void)ws_size;
    const int*   captions = (const int*)d_in[1];
    const float* embed_W  = (const float*)d_in[2];
    const float* W_ih_0   = (const float*)d_in[3];
    const float* W_hh_0   = (const float*)d_in[4];
    const float* b_ih_0   = (const float*)d_in[5];
    const float* b_hh_0   = (const float*)d_in[6];
    const float* W_ih_1   = (const float*)d_in[7];
    const float* W_hh_1   = (const float*)d_in[8];
    const float* b_ih_1   = (const float*)d_in[9];
    const float* b_hh_1   = (const float*)d_in[10];
    const float* fc_W     = (const float*)d_in[11];
    const float* fc_b     = (const float*)d_in[12];
    float* out = (float*)d_out;

    const int V = 50257, G = 2048, K = 512, M = 4032, T = 63;

    char* ws = (char*)d_ws;
    size_t off = 0;
    auto carve = [&](size_t bytes) -> void* {
        void* p = ws + off;
        off = (off + bytes + 255) & ~(size_t)255;
        return p;
    };
    __bf16* fcWbf  = (__bf16*)carve((size_t)V * K * 2);
    __bf16* Wih0bf = (__bf16*)carve((size_t)G * K * 2);
    __bf16* Whh0bf = (__bf16*)carve((size_t)G * K * 2);
    __bf16* Wih1bf = (__bf16*)carve((size_t)G * K * 2);
    __bf16* Whh1bf = (__bf16*)carve((size_t)G * K * 2);
    __bf16* Xbf    = (__bf16*)carve((size_t)M * K * 2);
    float*  xg0    = (float*)carve((size_t)M * G * 4);
    __bf16* h2     = (__bf16*)carve((size_t)(T + 1) * 64 * K * 2);  // slot t+1 = h2_t; slot 0 = zeros
    __bf16* h0     = (__bf16*)carve((size_t)2 * 64 * K * 2);        // ping-pong
    float*  c0     = (float*)carve((size_t)64 * K * 4);
    float*  c1     = (float*)carve((size_t)64 * K * 4);
    float*  bias0  = (float*)carve((size_t)G * 4);
    float*  bias1  = (float*)carve((size_t)G * 4);

    auto blks = [](int n) { return (n + 255) / 256; };

    // Weight conversion + bias folding + state init
    cvt_bf16_kernel<<<blks(V * K), 256, 0, stream>>>(fc_W,   fcWbf,  V * K);
    cvt_bf16_kernel<<<blks(G * K), 256, 0, stream>>>(W_ih_0, Wih0bf, G * K);
    cvt_bf16_kernel<<<blks(G * K), 256, 0, stream>>>(W_hh_0, Whh0bf, G * K);
    cvt_bf16_kernel<<<blks(G * K), 256, 0, stream>>>(W_ih_1, Wih1bf, G * K);
    cvt_bf16_kernel<<<blks(G * K), 256, 0, stream>>>(W_hh_1, Whh1bf, G * K);
    bias_sum_kernel<<<blks(G), 256, 0, stream>>>(b_ih_0, b_hh_0, bias0, G);
    bias_sum_kernel<<<blks(G), 256, 0, stream>>>(b_ih_1, b_hh_1, bias1, G);
    zero_f32_kernel<<<blks(64 * K), 256, 0, stream>>>(c0, 64 * K);
    zero_f32_kernel<<<blks(64 * K), 256, 0, stream>>>(c1, 64 * K);
    zero_bf16_kernel<<<blks(2 * 64 * K), 256, 0, stream>>>(h0, 2 * 64 * K);
    zero_bf16_kernel<<<blks(64 * K), 256, 0, stream>>>(h2, 64 * K);  // slot 0 only

    // Embedding gather (m = t*64 + b ordering)
    gather_embed_kernel<<<M, 256, 0, stream>>>(captions, embed_W, Xbf);

    // All layer-0 input gates in one big WMMA GEMM
    gemm_xg0_kernel<<<dim3(G / 128, M / 64), 256, 0, stream>>>(Xbf, Wih0bf, bias0, xg0);

    // Recurrent steps: 2 fused kernels per timestep, serialized on the stream
    for (int t = 0; t < T; ++t) {
        const __bf16* h0_prev = h0 + (size_t)(t & 1) * 64 * K;
        __bf16*       h0_new  = h0 + (size_t)((t + 1) & 1) * 64 * K;
        lstm_step_kernel<<<8, 256, 0, stream>>>(
            xg0 + (size_t)t * 64 * G, nullptr,
            nullptr, nullptr,
            h0_prev, Whh0bf, c0, h0_new);
        lstm_step_kernel<<<8, 256, 0, stream>>>(
            nullptr, bias1,
            h0_new, Wih1bf,
            h2 + (size_t)t * 64 * K, Whh1bf,
            c1, h2 + (size_t)(t + 1) * 64 * K);
    }

    // Final vocab projection (dominant 208 GFLOP GEMM)
    gemm_fc_kernel<<<dim3((V + 127) / 128, M / 64), 256, 0, stream>>>(
        h2 + (size_t)64 * K, fcWbf, fc_b, out, V);
}
